// SparseMoE_22789096473339
// MI455X (gfx1250) — matrix operand order repaired and checked by reading, not compiled, not run on hardware
//
#include <hip/hip_runtime.h>
#include <hip/hip_bf16.h>

#define MOE_D 1024
#define MOE_H 2048
#define MOE_E 8
#define MOE_BT 32   // token tile: two 16-row M-subtiles per wave

typedef __attribute__((ext_vector_type(16))) __bf16 bf16x16;
typedef __attribute__((ext_vector_type(8)))  __bf16 bf16x8;
typedef __attribute__((ext_vector_type(8)))  float  v8f;
typedef __attribute__((ext_vector_type(4)))  unsigned int u32x4;
typedef __attribute__((ext_vector_type(8)))  unsigned int u32x8;

#define WMMA_BF16(a, b, c) \
  __builtin_amdgcn_wmma_f32_16x16x32_bf16(false, (a), false, (b), (short)0, (c), false, false)

__device__ __forceinline__ unsigned short f32_to_bf16(float f) {
  union { float f; unsigned int u; } v; v.f = f;
  unsigned int r = v.u + 0x7FFFu + ((v.u >> 16) & 1u);
  return (unsigned short)(r >> 16);
}

__device__ __forceinline__ float fast_silu(float z) {
  // z * 1/(1+e^-z); v_rcp_f32 accuracy (~1ulp) far exceeds bf16 storage precision
  return z * __builtin_amdgcn_rcpf(1.0f + __expf(-z));
}

__device__ __forceinline__ bf16x16 load_afrag(const unsigned short* p) {
  bf16x8 lo = *(const bf16x8*)(p);
  bf16x8 hi = *(const bf16x8*)(p + 8);
  bf16x16 a;
#pragma unroll
  for (int i = 0; i < 8; ++i) { a[i] = lo[i]; a[i + 8] = hi[i]; }
  return a;
}

// ---------------- zero init for atomic accumulation ----------------
__global__ void zero_f32(float* __restrict__ p, size_t n) {
  size_t stride = (size_t)gridDim.x * blockDim.x;
  for (size_t i = (size_t)blockIdx.x * blockDim.x + threadIdx.x; i < n; i += stride)
    p[i] = 0.0f;
}

// ---------------- fp32 -> bf16 (RNE), 4-wide ----------------
__global__ void cvt_f32_bf16(const float* __restrict__ s, unsigned short* __restrict__ d, size_t n4) {
  size_t stride = (size_t)gridDim.x * blockDim.x;
  for (size_t i = (size_t)blockIdx.x * blockDim.x + threadIdx.x; i < n4; i += stride) {
    float4 f = ((const float4*)s)[i];
    ushort4 o;
    o.x = f32_to_bf16(f.x); o.y = f32_to_bf16(f.y);
    o.z = f32_to_bf16(f.z); o.w = f32_to_bf16(f.w);
    ((ushort4*)d)[i] = o;
  }
}

// ---------------- noisy top-2 router (full fp32), one wave32 per token ----------------
__global__ __launch_bounds__(256) void router_kernel(
    const float* __restrict__ x, const float* __restrict__ noise,
    const float* __restrict__ rw, const float* __restrict__ rb,
    const float* __restrict__ nw, const float* __restrict__ nb,
    float* __restrict__ gates, int T)
{
  constexpr int D = MOE_D, E = MOE_E;
  const int wave = threadIdx.x >> 5;
  const int lane = threadIdx.x & 31;
  const int t = blockIdx.x * 8 + wave;
  if (t >= T) return;

  const float* xr = x + (size_t)t * D;
  float accR[E], accN[E];
#pragma unroll
  for (int e = 0; e < E; ++e) { accR[e] = 0.0f; accN[e] = 0.0f; }

  for (int k = lane; k < D; k += 32) {
    float xv = xr[k];
#pragma unroll
    for (int e = 0; e < E; ++e) {
      accR[e] = fmaf(xv, rw[(size_t)e * D + k], accR[e]);
      accN[e] = fmaf(xv, nw[(size_t)e * D + k], accN[e]);
    }
  }
#pragma unroll
  for (int e = 0; e < E; ++e) {
#pragma unroll
    for (int off = 16; off > 0; off >>= 1) {
      accR[e] += __shfl_xor(accR[e], off, 32);
      accN[e] += __shfl_xor(accN[e], off, 32);
    }
  }

  if (lane == 0) {
    float noisy[E];
#pragma unroll
    for (int e = 0; e < E; ++e) {
      float lg = accR[e] + rb[e];
      float nl = accN[e] + nb[e];
      float sp = (nl > 20.0f) ? nl : log1pf(__expf(nl));
      noisy[e] = lg + noise[(size_t)t * E + e] * sp;
    }
    int i1 = 0;
#pragma unroll
    for (int j = 1; j < E; ++j) if (noisy[j] > noisy[i1]) i1 = j;
    int i2 = (i1 == 0) ? 1 : 0;
#pragma unroll
    for (int j = 0; j < E; ++j)
      if (j != i1 && j != i2 && noisy[j] > noisy[i2]) i2 = j;
    float ex = __expf(noisy[i2] - noisy[i1]);
    float g1 = 1.0f / (1.0f + ex);
    float g2 = ex / (1.0f + ex);
#pragma unroll
    for (int j = 0; j < E; ++j)
      gates[(size_t)t * E + j] = (j == i1) ? g1 : ((j == i2) ? g2 : 0.0f);
  }
}

// ---------------- fused SwiGLU expert, bf16 WMMA + TDM staging ----------------
__global__ __launch_bounds__(256) void moe_expert_kernel(
    const unsigned short* __restrict__ xb,
    const unsigned short* __restrict__ w1b,
    const unsigned short* __restrict__ w3b,
    const unsigned short* __restrict__ w2b,
    const float* __restrict__ gates,
    float* __restrict__ out)
{
  constexpr int D = MOE_D, H = MOE_H, E = MOE_E, BT = MOE_BT;
  // x tile written by TDM with pad_interval=256 dwords, pad_amount=2 dwords:
  // row stride = 1024 + 8 halfs = 1032, with +4 halfs inserted at k==512.
  constexpr int XROW = D + 8;
  constexpr int HP   = H + 8;          // hs row padding (manual), 4-bank stagger

  __shared__ __align__(16) unsigned short xs[BT * XROW];
  __shared__ __align__(16) unsigned short hs[BT][HP];
  __shared__ float gsh[BT];

  const int tid  = threadIdx.x;
  const int wave = tid >> 5;
  const int lane = tid & 31;
  const int t0   = blockIdx.x * BT;
  const int e    = blockIdx.y;

  // wave 0: load gates for this (tile, expert) and kick the TDM for the x tile
  if (tid < BT) gsh[tid] = gates[(size_t)(t0 + tid) * E + e];
  if (wave == 0) {
    const unsigned long long ga =
        (unsigned long long)(const void*)(xb + (size_t)t0 * D);
    const unsigned lds_off = (unsigned)(unsigned long long)(void*)&xs[0];

    u32x4 g0;
    g0[0] = 1u;                                              // count=1 valid D#
    g0[1] = lds_off;                                         // lds_addr
    g0[2] = (unsigned)(ga & 0xFFFFFFFFull);                  // global_addr[31:0]
    g0[3] = (unsigned)((ga >> 32) & 0x01FFFFFFull)           // global_addr[56:32]
            | (2u << 30);                                    // type=2 ("image")
    u32x8 g1;
    g1[0] = (1u << 16)      // data_size = 2 bytes (bf16)
          | (1u << 20)      // pad_enable
          | (7u << 22)      // pad_interval: 256 dwords
          | (1u << 25);     // pad_amount: 2 dwords (4 halfs)
    g1[1] = ((unsigned)D & 0xFFFFu) << 16;                   // tensor_dim0[15:0]
    g1[2] = ((unsigned)D >> 16) | ((unsigned)BT << 16);      // dim0 hi | tensor_dim1 lo
    g1[3] = ((unsigned)D << 16);                             // tile_dim0 = 1024
    g1[4] = (unsigned)BT;                                    // tile_dim1 = 32
    g1[5] = (unsigned)D;                                     // tensor_dim0_stride lo
    g1[6] = 0u;
    g1[7] = 0u;
    asm volatile(
        "tensor_load_to_lds %0, %1\n\t"
        "s_wait_tensorcnt 0x0"
        :: "s"(g0), "s"(g1)
        : "memory");
  }
  __syncthreads();

  bool any = false;
#pragma unroll
  for (int i = 0; i < BT; ++i) any = any || (gsh[i] != 0.0f);
  if (!any) return;   // uniform: whole tile unused by this expert

  // WMMA fragment lane geometry (ISA 7.12.2)
  const int arow = lane & 15;          // A: row M
  const int akb  = (lane >> 4) << 4;   // A: K base (16 contiguous K per lane)
  const int bkb  = (lane >> 4) << 4;   // B: K base (16 contiguous K per lane)
  const int bcol = lane & 15;          // B: column N
  const int mb   = (lane >> 4) << 3;   // C/D: row base

  const size_t woff = (size_t)e * H * D;
  const unsigned short* a0 = xs + arow * XROW + akb;
  const unsigned short* a1 = a0 + 16 * XROW;

  // ---- Stage 1: h = silu(x @ w1^T) * (x @ w3^T)  (2 M-tiles x 2 N-tiles / wave) ----
  for (int ctA = wave; ctA < 64; ctA += 8) {
    const int ctB = ctA + 64;
    v8f acc1[2][2] = {};   // [n-tile][m-tile]
    v8f acc3[2][2] = {};
    const unsigned short* b1pA = w1b + woff + (size_t)(ctA * 16 + bcol) * D + bkb;
    const unsigned short* b3pA = w3b + woff + (size_t)(ctA * 16 + bcol) * D + bkb;
    const unsigned short* b1pB = b1pA + (size_t)64 * 16 * D;
    const unsigned short* b3pB = b3pA + (size_t)64 * 16 * D;

    auto kstep = [&](int k0, int apad) {
      bf16x16 am0 = load_afrag(a0 + k0 + apad);
      bf16x16 am1 = load_afrag(a1 + k0 + apad);
      bf16x16 b1A = *(const bf16x16*)(b1pA + k0);
      bf16x16 b3A = *(const bf16x16*)(b3pA + k0);
      bf16x16 b1B = *(const bf16x16*)(b1pB + k0);
      bf16x16 b3B = *(const bf16x16*)(b3pB + k0);
      acc1[0][0] = WMMA_BF16(am0, b1A, acc1[0][0]);
      acc1[0][1] = WMMA_BF16(am1, b1A, acc1[0][1]);
      acc3[0][0] = WMMA_BF16(am0, b3A, acc3[0][0]);
      acc3[0][1] = WMMA_BF16(am1, b3A, acc3[0][1]);
      acc1[1][0] = WMMA_BF16(am0, b1B, acc1[1][0]);
      acc1[1][1] = WMMA_BF16(am1, b1B, acc1[1][1]);
      acc3[1][0] = WMMA_BF16(am0, b3B, acc3[1][0]);
      acc3[1][1] = WMMA_BF16(am1, b3B, acc3[1][1]);
    };
#pragma unroll 2
    for (int k0 = 0; k0 < 512; k0 += 32) kstep(k0, 0);
#pragma unroll 2
    for (int k0 = 512; k0 < 1024; k0 += 32) kstep(k0, 4);  // TDM mid-row pad

#pragma unroll
    for (int n = 0; n < 2; ++n) {
      const int col = ((n == 0) ? ctA : ctB) * 16 + bcol;
#pragma unroll
      for (int v = 0; v < 8; ++v) {
        hs[mb + v][col]      = f32_to_bf16(fast_silu(acc1[n][0][v]) * acc3[n][0][v]);
        hs[16 + mb + v][col] = f32_to_bf16(fast_silu(acc1[n][1][v]) * acc3[n][1][v]);
      }
    }
  }
  __syncthreads();

  // ---- Stage 2: out += gate * (h @ w2^T)  (2 M-tiles x 2 N-tiles / wave) ----
  const unsigned short* ha0 = &hs[arow][akb];
  const unsigned short* ha1 = &hs[arow + 16][akb];
  for (int dtA = wave; dtA < 32; dtA += 8) {
    const int dtB = dtA + 32;
    v8f acc[2][2] = {};    // [n-tile][m-tile]
    const unsigned short* bpA = w2b + (size_t)e * D * H + (size_t)(dtA * 16 + bcol) * H + bkb;
    const unsigned short* bpB = bpA + (size_t)32 * 16 * H;
#pragma unroll 4
    for (int k0 = 0; k0 < H; k0 += 32) {
      bf16x16 am0 = load_afrag(ha0 + k0);
      bf16x16 am1 = load_afrag(ha1 + k0);
      bf16x16 bA  = *(const bf16x16*)(bpA + k0);
      bf16x16 bB  = *(const bf16x16*)(bpB + k0);
      acc[0][0] = WMMA_BF16(am0, bA, acc[0][0]);
      acc[0][1] = WMMA_BF16(am1, bA, acc[0][1]);
      acc[1][0] = WMMA_BF16(am0, bB, acc[1][0]);
      acc[1][1] = WMMA_BF16(am1, bB, acc[1][1]);
    }
#pragma unroll
    for (int n = 0; n < 2; ++n) {
      const int col = ((n == 0) ? dtA : dtB) * 16 + bcol;
#pragma unroll
      for (int v = 0; v < 8; ++v) {
        float g0 = gsh[mb + v];
        float g1 = gsh[16 + mb + v];
        if (g0 != 0.0f)
          atomicAdd(out + (size_t)(t0 + mb + v) * D + col, acc[n][0][v] * g0);
        if (g1 != 0.0f)
          atomicAdd(out + (size_t)(t0 + 16 + mb + v) * D + col, acc[n][1][v] * g1);
      }
    }
  }
}

// ---------------- launcher ----------------
extern "C" void kernel_launch(void* const* d_in, const int* in_sizes, int n_in,
                              void* d_out, int out_size, void* d_ws, size_t ws_size,
                              hipStream_t stream) {
  const float* x     = (const float*)d_in[0];
  const float* noise = (const float*)d_in[1];
  const float* rw    = (const float*)d_in[2];
  const float* rb    = (const float*)d_in[3];
  const float* nw    = (const float*)d_in[4];
  const float* nb    = (const float*)d_in[5];
  const float* w1    = (const float*)d_in[6];
  const float* w2    = (const float*)d_in[7];
  const float* w3    = (const float*)d_in[8];
  float* out = (float*)d_out;

  const int D = MOE_D, H = MOE_H, E = MOE_E;
  const int T = in_sizes[0] / D;               // B*S
  const size_t wcount = (size_t)E * H * D;

  char* ws = (char*)d_ws;
  size_t off = 0;
  auto bump = [&](size_t bytes) -> void* {
    void* p = ws + off;
    off += bytes;
    off = (off + 255) & ~(size_t)255;
    return p;
  };
  unsigned short* xb    = (unsigned short*)bump((size_t)T * D * 2);
  unsigned short* w1b   = (unsigned short*)bump(wcount * 2);
  unsigned short* w3b   = (unsigned short*)bump(wcount * 2);
  unsigned short* w2b   = (unsigned short*)bump(wcount * 2);
  float*          gates = (float*)bump((size_t)T * E * 4);
  (void)ws_size; (void)n_in;

  zero_f32<<<2048, 256, 0, stream>>>(out, (size_t)out_size);
  cvt_f32_bf16<<<2048, 256, 0, stream>>>(x,  xb,  ((size_t)T * D) / 4);
  cvt_f32_bf16<<<4096, 256, 0, stream>>>(w1, w1b, wcount / 4);
  cvt_f32_bf16<<<4096, 256, 0, stream>>>(w3, w3b, wcount / 4);
  cvt_f32_bf16<<<4096, 256, 0, stream>>>(w2, w2b, wcount / 4);

  router_kernel<<<(T + 7) / 8, 256, 0, stream>>>(x, noise, rw, rb, nw, nb, gates, T);

  dim3 grid(T / MOE_BT, E);
  moe_expert_kernel<<<grid, 256, 0, stream>>>(xb, w1b, w3b, w2b, gates, out);
}
